// EncoderBlock_63264868270221
// MI455X (gfx1250) — matrix operand order repaired and checked by reading, not compiled, not run on hardware
//
#include <hip/hip_runtime.h>

// ---------------------------------------------------------------------------
// Infini-attention encoder block for MI455X (gfx1250, wave32, WMMA).
// All heavy matmuls use v_wmma_f32_16x16x32_f16 (f16 in, f32 accumulate).
// Round 3: transposed LDS staging now uses coalesced b128 global loads +
// ds_store_b16 scatter (was scalar global_load_u16); seg_kv K-step 64;
// GEMM adds global_prefetch_b8 for the tile after next.
// ---------------------------------------------------------------------------

#define B_    4
#define S_    4096
#define E_    1024
#define H_    16
#define D_    64
#define HD_   (H_ * D_)      // 1024
#define FFN_  4096
#define SEG_  512
#define NSEG_ 8
#define SCALE_ 0.125f        // 1/sqrt(64)
#define EPS_   1e-6f

typedef __attribute__((ext_vector_type(16))) _Float16 v16h;
typedef __attribute__((ext_vector_type(8)))  float    v8f;
typedef __attribute__((ext_vector_type(4)))  unsigned int u32x4;

union Frag16 { v16h v; u32x4 q[2]; };

__device__ __forceinline__ float sigmaf(float t) {  // elu(t)+1
  return t > 0.f ? t + 1.f : __expf(t);
}

__device__ __forceinline__ v8f wmma16(const Frag16& a, const Frag16& b, v8f c) {
  return __builtin_amdgcn_wmma_f32_16x16x32_f16(
      false, a.v, false, b.v, (short)0, c, false, false);
}

// ---------------------------------------------------------------------------
// f32 -> f16 conversion (activations)
// ---------------------------------------------------------------------------
__global__ void cvt_f32_f16_kernel(const float* __restrict__ in,
                                   _Float16* __restrict__ out, int n) {
  int i = blockIdx.x * 256 + threadIdx.x;
  if (i < n) out[i] = (_Float16)in[i];
}

// ---------------------------------------------------------------------------
// f32 [R][C] -> f16 [C][R] transposed conversion (weights), tiled via LDS so
// both global sides stay coalesced. R, C multiples of 32.
// ---------------------------------------------------------------------------
__global__ __launch_bounds__(256) void cvt_transpose_kernel(
    const float* __restrict__ in, _Float16* __restrict__ out, int R, int C) {
  __shared__ float tile[32][33];
  const int bx = blockIdx.x * 32;               // col base (C)
  const int by = blockIdx.y * 32;               // row base (R)
  const int tx = threadIdx.x & 31, ty = threadIdx.x >> 5;  // 32 x 8
#pragma unroll
  for (int j = 0; j < 32; j += 8)
    tile[ty + j][tx] = in[(size_t)(by + ty + j) * C + bx + tx];
  __syncthreads();
#pragma unroll
  for (int j = 0; j < 32; j += 8)
    out[(size_t)(bx + ty + j) * R + by + tx] = (_Float16)tile[tx][ty + j];
}

// ---------------------------------------------------------------------------
// sigma (elu+1) elementwise on f16 (precomputes sigma(K) so seg_kv's inner
// loop is pure WMMA, no transcendental traffic against the matrix pipe)
// ---------------------------------------------------------------------------
__global__ void sigma_f16_kernel(const _Float16* __restrict__ in,
                                 _Float16* __restrict__ out, int n) {
  int i = blockIdx.x * 256 + threadIdx.x;
  if (i < n) out[i] = (_Float16)sigmaf((float)in[i]);
}

// ---------------------------------------------------------------------------
// Generic WMMA GEMM: C[M,N] = A[M,K] * Bt[N,K]^T (+bias)(+ReLU), f16 in.
// B is pre-transposed ([N][K]) so both LDS stages are aligned b128 copies.
// Block: 256 threads (8 waves), tile 128x128, K-step 64 (2 WMMA k-steps per
// barrier). Next tile prefetched into registers during compute; tile after
// next warmed in L2 via global_prefetch_b8.
// M % 128 == 0, N % 128 == 0, K % 64 == 0 (true for all uses here).
// ---------------------------------------------------------------------------
template <bool OUT16, bool BIAS, bool RELU>
__global__ __launch_bounds__(256) void gemm_wmma_kernel(
    const _Float16* __restrict__ A, int lda,
    const _Float16* __restrict__ Bt, int ldb,
    float* __restrict__ Cf, _Float16* __restrict__ Ch, int ldc,
    const float* __restrict__ bias, int M, int N, int K) {
  constexpr int LK = 72;  // 64 + 8 halves pad; 144B row keeps b128 alignment
  __shared__ __align__(16) _Float16 As[128 * LK];
  __shared__ __align__(16) _Float16 Bs[128 * LK];

  const int tid   = threadIdx.x;
  const int wave  = tid >> 5;
  const int lane  = tid & 31;
  const int lhalf = lane & 15;
  const int wy = wave >> 2, wx = wave & 3;         // 2 x 4 wave grid
  const int m0 = blockIdx.y * 128, n0 = blockIdx.x * 128;
  const int kA = (lane < 16) ? 0 : 8;              // A frag chunk base (halves)
  const int kB = (lane < 16) ? 0 : 8;              // B frag chunk base (halves)
  const int rb = (lane < 16) ? 0 : 8;              // C row base

  // staging geometry: 128 rows x 64 halves = 1024 x 16B chunks, 4 per thread
  const int srow = tid >> 3;                       // + c*32
  const int scol = (tid & 7) * 8;

  v8f acc[4][2] = {};
  u32x4 ra[4], rbv[4];
#pragma unroll
  for (int c = 0; c < 4; ++c) {
    int r = srow + c * 32;
    ra[c]  = *(const u32x4*)(A  + (size_t)(m0 + r) * lda + scol);
    rbv[c] = *(const u32x4*)(Bt + (size_t)(n0 + r) * ldb + scol);
  }

  for (int k0 = 0; k0 < K; k0 += 64) {
    if (k0) __syncthreads();                       // protect LDS reuse
#pragma unroll
    for (int c = 0; c < 4; ++c) {
      int r = srow + c * 32;
      *(u32x4*)(&As[r * LK + scol]) = ra[c];
      *(u32x4*)(&Bs[r * LK + scol]) = rbv[c];
    }
    __syncthreads();
    if (k0 + 64 < K) {
#pragma unroll
      for (int c = 0; c < 4; ++c) {
        int r = srow + c * 32;
        ra[c]  = *(const u32x4*)(A  + (size_t)(m0 + r) * lda + k0 + 64 + scol);
        rbv[c] = *(const u32x4*)(Bt + (size_t)(n0 + r) * ldb + k0 + 64 + scol);
      }
    }
    if (k0 + 128 < K) {                            // L2 warm for tile+2
      __builtin_prefetch(A  + (size_t)(m0 + srow) * lda + k0 + 128 + scol, 0, 1);
      __builtin_prefetch(Bt + (size_t)(n0 + srow) * ldb + k0 + 128 + scol, 0, 1);
    }
#pragma unroll
    for (int ks = 0; ks < 2; ++ks) {
      Frag16 bf[2];
#pragma unroll
      for (int fn = 0; fn < 2; ++fn) {
        const _Float16* p = &Bs[(wx * 32 + fn * 16 + lhalf) * LK + ks * 32 + kB];
        bf[fn].q[0] = *(const u32x4*)(p);
        bf[fn].q[1] = *(const u32x4*)(p + 16);
      }
#pragma unroll
      for (int fm = 0; fm < 4; ++fm) {
        Frag16 af;
        const _Float16* p = &As[(wy * 64 + fm * 16 + lhalf) * LK + ks * 32 + kA];
        af.q[0] = *(const u32x4*)(p);
        af.q[1] = *(const u32x4*)(p + 16);
#pragma unroll
        for (int fn = 0; fn < 2; ++fn)
          acc[fm][fn] = wmma16(af, bf[fn], acc[fm][fn]);
      }
    }
  }

#pragma unroll
  for (int fm = 0; fm < 4; ++fm)
#pragma unroll
    for (int fn = 0; fn < 2; ++fn)
#pragma unroll
      for (int r = 0; r < 8; ++r) {
        int row = m0 + wy * 64 + fm * 16 + r + rb;
        int col = n0 + wx * 32 + fn * 16 + lhalf;
        float v = acc[fm][fn][r];
        if (BIAS) v += bias[col];
        if (RELU) v = fmaxf(v, 0.f);
        if (OUT16) Ch[(size_t)row * ldc + col] = (_Float16)v;
        else       Cf[(size_t)row * ldc + col] = v;
      }
}

// ---------------------------------------------------------------------------
// Per-segment compressive-memory sums (sigma(K) precomputed -> pure WMMA):
//   Mseg[b,h,seg] = sK_seg^T (64x512) @ V_seg (512x64)   (f32 64x64)
//   zseg[b,h,seg][d] = sum_t sK_seg[t][d]
// One block per (b,h,seg). K-step 64; staging is coalesced b128 global loads
// (8 consecutive lanes cover one 128B row) + ds_store_b16 transpose scatter.
// ---------------------------------------------------------------------------
__global__ __launch_bounds__(256) void seg_kv_kernel(
    const _Float16* __restrict__ sK16, const _Float16* __restrict__ V16,
    float* __restrict__ Mseg, float* __restrict__ zseg) {
  constexpr int LK = 72;
  __shared__ __align__(16) _Float16 Kt[64 * LK];   // Kt[d][t] = sK^T
  __shared__ __align__(16) _Float16 Vt[64 * LK];   // Vt[e][t] (B transposed)

  const int idx = blockIdx.x;                      // bh*NSEG + seg
  const int seg = idx & (NSEG_ - 1);
  const int bh  = idx >> 3;
  const int b = bh >> 4, h = bh & 15;
  const int tid = threadIdx.x, wave = tid >> 5, lane = tid & 31;
  const int lhalf = lane & 15;
  const int wy = wave >> 2, wx = wave & 3;         // M 2x32, N 4x16
  const int kA = (lane < 16) ? 0 : 8;
  const int kB = (lane < 16) ? 0 : 8;
  const int rb = (lane < 16) ? 0 : 8;

  const size_t base = ((size_t)b * S_ + seg * SEG_) * HD_ + h * D_;
  v8f acc[2] = {};
  float zacc = 0.f;

  for (int k0 = 0; k0 < SEG_; k0 += 64) {
    __syncthreads();
#pragma unroll
    for (int i = 0; i < 2; ++i) {                  // 512 chunks of 8 halves
      int c = tid + i * 256;
      int t = c >> 3, d8 = (c & 7) * 8;            // lanes sweep a full row
      u32x4 dk = *(const u32x4*)(sK16 + base + (size_t)(k0 + t) * HD_ + d8);
      u32x4 dv = *(const u32x4*)(V16  + base + (size_t)(k0 + t) * HD_ + d8);
      const _Float16* hk = (const _Float16*)&dk;
      const _Float16* hv = (const _Float16*)&dv;
#pragma unroll
      for (int j = 0; j < 8; ++j) {
        Kt[(d8 + j) * LK + t] = hk[j];
        Vt[(d8 + j) * LK + t] = hv[j];
      }
    }
    __syncthreads();
    if (tid < 64) {                                // z partial: b128 LDS reads
#pragma unroll
      for (int t8 = 0; t8 < 64; t8 += 8) {
        u32x4 dk = *(const u32x4*)(&Kt[tid * LK + t8]);
        const _Float16* hk = (const _Float16*)&dk;
#pragma unroll
        for (int j = 0; j < 8; ++j) zacc += (float)hk[j];
      }
    }
#pragma unroll
    for (int ks = 0; ks < 2; ++ks) {
      Frag16 bf;
      const _Float16* pv = &Vt[(wx * 16 + lhalf) * LK + ks * 32 + kB];
      bf.q[0] = *(const u32x4*)(pv);
      bf.q[1] = *(const u32x4*)(pv + 16);
#pragma unroll
      for (int fm = 0; fm < 2; ++fm) {
        Frag16 af;
        const _Float16* pk = &Kt[(wy * 32 + fm * 16 + lhalf) * LK + ks * 32 + kA];
        af.q[0] = *(const u32x4*)(pk);
        af.q[1] = *(const u32x4*)(pk + 16);
        acc[fm] = wmma16(af, bf, acc[fm]);
      }
    }
  }

  float* Mb = Mseg + (size_t)idx * (D_ * D_);
#pragma unroll
  for (int fm = 0; fm < 2; ++fm)
#pragma unroll
    for (int r = 0; r < 8; ++r)
      Mb[(wy * 32 + fm * 16 + r + rb) * D_ + wx * 16 + lhalf] = acc[fm][r];
  if (tid < 64) zseg[(size_t)idx * D_ + tid] = zacc;
}

// ---------------------------------------------------------------------------
// Exclusive prefix scans over the 8 segments (+ inclusive total = context).
// MpreT stored transposed ([e][d], f16) = WMMA B-operand layout for retrieval.
// ---------------------------------------------------------------------------
__global__ void mscan_kernel(const float* __restrict__ Mseg,
                             _Float16* __restrict__ MpreT,
                             float* __restrict__ ctx) {
  int i = blockIdx.x * 256 + threadIdx.x;          // over B*H*D*D
  int de = i & 4095, bh = i >> 12;
  int d = de >> 6, e = de & 63;
  float run = 0.f;
#pragma unroll
  for (int s = 0; s < NSEG_; ++s) {
    size_t o = (size_t)(bh * NSEG_ + s) * (D_ * D_);
    MpreT[o + e * D_ + d] = (_Float16)run;
    run += Mseg[o + d * D_ + e];
  }
  ctx[(size_t)bh * (D_ * D_) + d * D_ + e] = run;
}

__global__ void zscan_kernel(const float* __restrict__ zseg,
                             float* __restrict__ zpre) {
  int i = blockIdx.x * 256 + threadIdx.x;          // over B*H*D
  int d = i & 63, bh = i >> 6;
  float run = 0.f;
#pragma unroll
  for (int s = 0; s < NSEG_; ++s) {
    size_t o = (size_t)(bh * NSEG_ + s) * D_ + d;
    zpre[o] = run;
    run += zseg[o];
  }
}

// ---------------------------------------------------------------------------
// Fused infini-attention: one block per (b,h,seg, 128-query tile).
// 8 waves; wave w owns query rows [16w,16w+16) -> per-row softmax stats stay
// wave-local (shfl reductions within 16-lane halves). Flash-style online
// softmax over 8 key tiles of 64; P relayout C->A via LDS. Memory-retrieval
// term fused with two more WMMA passes; gated combine written as f16.
// ---------------------------------------------------------------------------
__global__ __launch_bounds__(256) void attn_kernel(
    const _Float16* __restrict__ Q16, const _Float16* __restrict__ K16,
    const _Float16* __restrict__ V16, const _Float16* __restrict__ MpreT,
    const float* __restrict__ zpre, const float* __restrict__ beta,
    _Float16* __restrict__ attn16) {
  constexpr int LQ = 72;  // 64 + 8 halves pad
  __shared__ __align__(16) _Float16 Qs[128 * LQ];
  __shared__ __align__(16) _Float16 Ks[64 * LQ];   // B^T for scores = K rows
  __shared__ __align__(16) _Float16 Vt[64 * LQ];   // Vt[e][t]
  __shared__ __align__(16) _Float16 Ps[128 * LQ];  // softmax probs (A relayout)
  __shared__ float zs[64];
  __shared__ float denbuf[8][16];

  const int bh  = blockIdx.z;
  const int b = bh >> 4, h = bh & 15;
  const int seg = blockIdx.y;
  const int q0  = blockIdx.x * 128;
  const int tid = threadIdx.x, wave = tid >> 5, lane = tid & 31;
  const int lhalf = lane & 15;
  const int kA = (lane < 16) ? 0 : 8;
  const int kB = (lane < 16) ? 0 : 8;
  const int rb = (lane < 16) ? 0 : 8;

  const size_t base = ((size_t)b * S_ + seg * SEG_) * HD_ + h * D_;

  // stage Q tile (128 x 64) and zpre
#pragma unroll
  for (int c = 0; c < 4; ++c) {
    int chunk = tid + c * 256;                     // row(128) x col8(8)
    int r = chunk >> 3, cc = (chunk & 7) * 8;
    u32x4 d = *(const u32x4*)(Q16 + base + (size_t)(q0 + r) * HD_ + cc);
    *(u32x4*)(&Qs[r * LQ + cc]) = d;
  }
  if (tid < 64) zs[tid] = zpre[(size_t)(bh * NSEG_ + seg) * D_ + tid];
  __syncthreads();

  // preload Q A-fragments (2 k-steps over D=64), reused for all key tiles
  Frag16 aQ[2];
  {
    const _Float16* p = &Qs[(wave * 16 + lhalf) * LQ];
#pragma unroll
    for (int ks = 0; ks < 2; ++ks) {
      aQ[ks].q[0] = *(const u32x4*)(p + ks * 32 + kA);
      aQ[ks].q[1] = *(const u32x4*)(p + ks * 32 + 16 + kA);
    }
  }
  const float gate = 1.f / (1.f + __expf(-beta[h]));

  v8f acc[4] = {};
  float m_run[8], l_run[8];
#pragma unroll
  for (int r = 0; r < 8; ++r) { m_run[r] = -1e30f; l_run[r] = 0.f; }

  for (int kt = 0; kt < SEG_ / 64; ++kt) {
    __syncthreads();
#pragma unroll
    for (int c = 0; c < 2; ++c) {                  // K tile 64x64 (row-major)
      int chunk = tid + c * 256;
      int r = chunk >> 3, cc = (chunk & 7) * 8;
      u32x4 d = *(const u32x4*)(K16 + base + (size_t)(kt * 64 + r) * HD_ + cc);
      *(u32x4*)(&Ks[r * LQ + cc]) = d;
    }
#pragma unroll
    for (int i = 0; i < 2; ++i) {                  // V tile transposed stage
      int c = tid + i * 256;
      int t = c >> 3, d8 = (c & 7) * 8;            // lanes sweep a full row
      u32x4 dv = *(const u32x4*)(V16 + base + (size_t)(kt * 64 + t) * HD_ + d8);
      const _Float16* hv = (const _Float16*)&dv;
#pragma unroll
      for (int j = 0; j < 8; ++j) Vt[(d8 + j) * LQ + t] = hv[j];
    }
    __syncthreads();

    // scores S = Q K^T (16 rows x 64 keys per wave)
    v8f sfr[4];
#pragma unroll
    for (int fn = 0; fn < 4; ++fn) {
      sfr[fn] = (v8f){};
#pragma unroll
      for (int ks = 0; ks < 2; ++ks) {
        Frag16 bf;
        const _Float16* p = &Ks[(fn * 16 + lhalf) * LQ + ks * 32 + kB];
        bf.q[0] = *(const u32x4*)(p);
        bf.q[1] = *(const u32x4*)(p + 16);
        sfr[fn] = wmma16(aQ[ks], bf, sfr[fn]);
      }
    }

    // online softmax update (per-row stats via shfl within 16-lane halves)
#pragma unroll
    for (int r = 0; r < 8; ++r) {
      float mx = -1e30f;
#pragma unroll
      for (int fn = 0; fn < 4; ++fn) {
        float t = sfr[fn][r] * SCALE_;
        sfr[fn][r] = t;
        mx = fmaxf(mx, t);
      }
#pragma unroll
      for (int msk = 1; msk < 16; msk <<= 1)
        mx = fmaxf(mx, __shfl_xor(mx, msk, 32));
      float mnew = fmaxf(m_run[r], mx);
      float alpha = __expf(m_run[r] - mnew);
      float rs = 0.f;
#pragma unroll
      for (int fn = 0; fn < 4; ++fn) {
        float pv = __expf(sfr[fn][r] - mnew);
        sfr[fn][r] = pv;
        rs += pv;
      }
#pragma unroll
      for (int msk = 1; msk < 16; msk <<= 1) rs += __shfl_xor(rs, msk, 32);
      l_run[r] = l_run[r] * alpha + rs;
      m_run[r] = mnew;
#pragma unroll
      for (int fn2 = 0; fn2 < 4; ++fn2) acc[fn2][r] *= alpha;
    }

    // relayout P: C-layout -> LDS -> A-layout
    const int prow = wave * 16;
#pragma unroll
    for (int fn = 0; fn < 4; ++fn)
#pragma unroll
      for (int r = 0; r < 8; ++r)
        Ps[(prow + r + rb) * LQ + fn * 16 + lhalf] = (_Float16)sfr[fn][r];
    __syncthreads();

    // acc += P @ V
#pragma unroll
    for (int ks = 0; ks < 2; ++ks) {
      Frag16 aP;
      const _Float16* p = &Ps[(prow + lhalf) * LQ];
      aP.q[0] = *(const u32x4*)(p + ks * 32 + kA);
      aP.q[1] = *(const u32x4*)(p + ks * 32 + 16 + kA);
#pragma unroll
      for (int fn2 = 0; fn2 < 4; ++fn2) {
        Frag16 bf;
        const _Float16* pv = &Vt[(fn2 * 16 + lhalf) * LQ + ks * 32 + kB];
        bf.q[0] = *(const u32x4*)(pv);
        bf.q[1] = *(const u32x4*)(pv + 16);
        acc[fn2] = wmma16(aP, bf, acc[fn2]);
      }
    }
  }

  // memory retrieval numerator: sigma(Q) @ Mpre (B frags straight from HBM;
  // MpreT rows are 128B so every 16B chunk is aligned)
  Frag16 aS[2];
#pragma unroll
  for (int ks = 0; ks < 2; ++ks) {
    aS[ks] = aQ[ks];
#pragma unroll
    for (int i = 0; i < 16; ++i)
      aS[ks].v[i] = (_Float16)sigmaf((float)aS[ks].v[i]);
  }
  const _Float16* Mb = MpreT + (size_t)(bh * NSEG_ + seg) * (D_ * D_);
  v8f num[4] = {};
#pragma unroll
  for (int fn = 0; fn < 4; ++fn)
#pragma unroll
    for (int ks = 0; ks < 2; ++ks) {
      Frag16 bf;
      const _Float16* p = Mb + (fn * 16 + lhalf) * D_ + ks * 32 + kB;
      bf.q[0] = *(const u32x4*)(p);
      bf.q[1] = *(const u32x4*)(p + 16);
      num[fn] = wmma16(aS[ks], bf, num[fn]);
    }

  // denominator: sigma(q_row) . zpre  (one row per lane 0..15)
  if (lane < 16) {
    int row = wave * 16 + lane;
    float ds = 0.f;
#pragma unroll
    for (int d = 0; d < D_; ++d) ds += sigmaf((float)Qs[row * LQ + d]) * zs[d];
    denbuf[wave][lane] = ds + EPS_;
  }
  __syncthreads();

  // gated combine + store f16 [B,S,H,D]
#pragma unroll
  for (int fn = 0; fn < 4; ++fn)
#pragma unroll
    for (int r = 0; r < 8; ++r) {
      int rr = r + rb;
      float am = num[fn][r] / denbuf[wave][rr];
      float ad = acc[fn][r] / l_run[r];
      float o  = gate * am + (1.f - gate) * ad;
      int srow = seg * SEG_ + q0 + wave * 16 + rr;
      attn16[((size_t)b * S_ + srow) * HD_ + h * D_ + fn * 16 + lhalf] =
          (_Float16)o;
    }
}

// ---------------------------------------------------------------------------
// LayerNorm(x + y) -> f32 (and optional f16 copy). One block per token row.
// ---------------------------------------------------------------------------
__global__ __launch_bounds__(256) void ln_kernel(
    const float* __restrict__ X, const float* __restrict__ Y,
    const float* __restrict__ g, const float* __restrict__ bb,
    float* __restrict__ out32, _Float16* __restrict__ out16) {
  __shared__ float ws1[8], ws2[8];
  const int row = blockIdx.x, tid = threadIdx.x;
  const int wave = tid >> 5, lane = tid & 31;
  const float* xr = X + (size_t)row * E_;
  const float* yr = Y + (size_t)row * E_;
  float v[4], s = 0.f, s2 = 0.f;
#pragma unroll
  for (int j = 0; j < 4; ++j) {
    int i = tid + j * 256;
    float t = xr[i] + yr[i];
    v[j] = t; s += t; s2 += t * t;
  }
#pragma unroll
  for (int msk = 1; msk < 32; msk <<= 1) {
    s  += __shfl_xor(s,  msk, 32);
    s2 += __shfl_xor(s2, msk, 32);
  }
  if (lane == 0) { ws1[wave] = s; ws2[wave] = s2; }
  __syncthreads();
  float ts = 0.f, ts2 = 0.f;
#pragma unroll
  for (int w = 0; w < 8; ++w) { ts += ws1[w]; ts2 += ws2[w]; }
  float mu  = ts / (float)E_;
  float var = ts2 / (float)E_ - mu * mu;
  float rstd = rsqrtf(var + 1e-5f);
#pragma unroll
  for (int j = 0; j < 4; ++j) {
    int i = tid + j * 256;
    float t = (v[j] - mu) * rstd * g[i] + bb[i];
    out32[(size_t)row * E_ + i] = t;
    if (out16) out16[(size_t)row * E_ + i] = (_Float16)t;
  }
}

// ---------------------------------------------------------------------------
extern "C" void kernel_launch(void* const* d_in, const int* in_sizes, int n_in,
                              void* d_out, int out_size, void* d_ws,
                              size_t ws_size, hipStream_t stream) {
  (void)in_sizes; (void)n_in; (void)out_size; (void)ws_size;
  const float* x    = (const float*)d_in[0];
  const float* wq   = (const float*)d_in[1];
  const float* wk   = (const float*)d_in[2];
  const float* wv   = (const float*)d_in[3];
  const float* wo   = (const float*)d_in[4];
  const float* beta = (const float*)d_in[5];
  const float* ln1g = (const float*)d_in[6];
  const float* ln1b = (const float*)d_in[7];
  const float* w1   = (const float*)d_in[8];
  const float* b1   = (const float*)d_in[9];
  const float* w2   = (const float*)d_in[10];
  const float* b2   = (const float*)d_in[11];
  const float* ln2g = (const float*)d_in[12];
  const float* ln2b = (const float*)d_in[13];

  float* y_out   = (float*)d_out;
  float* ctx_out = y_out + (size_t)B_ * S_ * E_;

  const size_t NTOK = (size_t)B_ * S_;
  char* cur = (char*)d_ws;
  auto alloc = [&](size_t bytes) {
    void* r = (void*)cur;
    cur += (bytes + 255) & ~(size_t)255;
    return r;
  };
  _Float16* x16   = (_Float16*)alloc(NTOK * E_ * 2);
  _Float16* wq16t = (_Float16*)alloc((size_t)E_ * HD_ * 2);   // [HD][E]
  _Float16* wk16t = (_Float16*)alloc((size_t)E_ * HD_ * 2);
  _Float16* wv16t = (_Float16*)alloc((size_t)E_ * HD_ * 2);
  _Float16* wo16t = (_Float16*)alloc((size_t)HD_ * E_ * 2);   // [E][HD]
  _Float16* w116t = (_Float16*)alloc((size_t)E_ * FFN_ * 2);  // [FFN][E]
  _Float16* w216t = (_Float16*)alloc((size_t)FFN_ * E_ * 2);  // [E][FFN]
  _Float16* Q16   = (_Float16*)alloc(NTOK * HD_ * 2);
  _Float16* K16   = (_Float16*)alloc(NTOK * HD_ * 2);
  _Float16* V16   = (_Float16*)alloc(NTOK * HD_ * 2);
  _Float16* sK16  = (_Float16*)alloc(NTOK * HD_ * 2);
  _Float16* attn16= (_Float16*)alloc(NTOK * HD_ * 2);
  float*    Mseg  = (float*)alloc((size_t)B_ * H_ * NSEG_ * D_ * D_ * 4);
  float*    zseg  = (float*)alloc((size_t)B_ * H_ * NSEG_ * D_ * 4);
  _Float16* MpreT = (_Float16*)alloc((size_t)B_ * H_ * NSEG_ * D_ * D_ * 2);
  float*    zpre  = (float*)alloc((size_t)B_ * H_ * NSEG_ * D_ * 4);
  float*    aproj = (float*)alloc(NTOK * E_ * 4);
  float*    h32   = (float*)alloc(NTOK * E_ * 4);
  _Float16* h16   = (_Float16*)alloc(NTOK * E_ * 2);
  _Float16* ffn1  = (_Float16*)alloc(NTOK * FFN_ * 2);
  float*    ffn2o = (float*)alloc(NTOK * E_ * 4);

  const dim3 blk(256);
  {
    size_t n = NTOK * E_;
    cvt_f32_f16_kernel<<<dim3((unsigned)((n + 255) / 256)), blk, 0, stream>>>(
        x, x16, (int)n);
  }
  // weights: f32 [R][C] -> f16 [C][R]
  cvt_transpose_kernel<<<dim3(HD_/32, E_/32), blk, 0, stream>>>(wq, wq16t, E_, HD_);
  cvt_transpose_kernel<<<dim3(HD_/32, E_/32), blk, 0, stream>>>(wk, wk16t, E_, HD_);
  cvt_transpose_kernel<<<dim3(HD_/32, E_/32), blk, 0, stream>>>(wv, wv16t, E_, HD_);
  cvt_transpose_kernel<<<dim3(E_/32, HD_/32), blk, 0, stream>>>(wo, wo16t, HD_, E_);
  cvt_transpose_kernel<<<dim3(FFN_/32, E_/32), blk, 0, stream>>>(w1, w116t, E_, FFN_);
  cvt_transpose_kernel<<<dim3(E_/32, FFN_/32), blk, 0, stream>>>(w2, w216t, FFN_, E_);

  // Q/K/V projections (f16 out)
  gemm_wmma_kernel<true, false, false>
      <<<dim3(HD_ / 128, (unsigned)(NTOK / 128)), blk, 0, stream>>>(
          x16, E_, wq16t, E_, nullptr, Q16, HD_, nullptr, (int)NTOK, HD_, E_);
  gemm_wmma_kernel<true, false, false>
      <<<dim3(HD_ / 128, (unsigned)(NTOK / 128)), blk, 0, stream>>>(
          x16, E_, wk16t, E_, nullptr, K16, HD_, nullptr, (int)NTOK, HD_, E_);
  gemm_wmma_kernel<true, false, false>
      <<<dim3(HD_ / 128, (unsigned)(NTOK / 128)), blk, 0, stream>>>(
          x16, E_, wv16t, E_, nullptr, V16, HD_, nullptr, (int)NTOK, HD_, E_);

  // sigma(K) once (keeps transcendentals out of the WMMA inner loops)
  {
    size_t n = NTOK * HD_;
    sigma_f16_kernel<<<dim3((unsigned)((n + 255) / 256)), blk, 0, stream>>>(
        K16, sK16, (int)n);
  }

  // compressive-memory segment sums + prefix scans (context = inclusive total)
  seg_kv_kernel<<<dim3(B_ * H_ * NSEG_), blk, 0, stream>>>(sK16, V16, Mseg, zseg);
  mscan_kernel<<<dim3((B_ * H_ * D_ * D_) / 256), blk, 0, stream>>>(
      Mseg, MpreT, ctx_out);
  zscan_kernel<<<dim3((B_ * H_ * D_) / 256), blk, 0, stream>>>(zseg, zpre);

  // fused flash attention + memory retrieval + gate
  attn_kernel<<<dim3(SEG_ / 128, NSEG_, B_ * H_), blk, 0, stream>>>(
      Q16, K16, V16, MpreT, zpre, beta, attn16);

  // output projection (f32 out)
  gemm_wmma_kernel<false, false, false>
      <<<dim3(E_ / 128, (unsigned)(NTOK / 128)), blk, 0, stream>>>(
          attn16, HD_, wo16t, HD_, aproj, nullptr, E_, nullptr, (int)NTOK, E_, HD_);

  // h = LN(x + attn_out)
  ln_kernel<<<dim3((unsigned)NTOK), blk, 0, stream>>>(x, aproj, ln1g, ln1b,
                                                      h32, h16);
  // FFN
  gemm_wmma_kernel<true, true, true>
      <<<dim3(FFN_ / 128, (unsigned)(NTOK / 128)), blk, 0, stream>>>(
          h16, E_, w116t, E_, nullptr, ffn1, FFN_, b1, (int)NTOK, FFN_, E_);
  gemm_wmma_kernel<false, true, false>
      <<<dim3(E_ / 128, (unsigned)(NTOK / 128)), blk, 0, stream>>>(
          ffn1, FFN_, w216t, FFN_, ffn2o, nullptr, E_, b2, (int)NTOK, E_, FFN_);

  // y = LN(h + ffn)
  ln_kernel<<<dim3((unsigned)NTOK), blk, 0, stream>>>(h32, ffn2o, ln2g, ln2b,
                                                      y_out, nullptr);
}
